// GLMMoE_V2_5231270167124
// MI455X (gfx1250) — matrix operand order repaired and checked by reading, not compiled, not run on hardware
//
#include <hip/hip_runtime.h>
#include <hip/hip_bf16.h>
#include <stdint.h>

#define T_TOK 8192
#define HID   1024
#define NE    8
#define TOPK  2
#define IEXP  512
#define NSLOT (T_TOK * TOPK)

typedef __attribute__((ext_vector_type(16))) _Float16 v16h;
typedef __attribute__((ext_vector_type(8)))  float    v8f;
typedef __attribute__((ext_vector_type(4)))  _Float16 h4;
typedef __attribute__((ext_vector_type(2)))  _Float16 h2;

union FragH { v16h v; uint4 q[2]; };

#define BM 64
#define BN 128   // output cols per block
#define BK 32
#define LDA 40   // LDS row stride in halves (80B rows -> 16B-aligned fragment loads)
#define LDB 40

// ---------------------------------------------------------------- router
__global__ __launch_bounds__(256) void router_kernel(
    const float* __restrict__ X, const float* __restrict__ Wg,
    int* __restrict__ tok_e, float* __restrict__ tok_w, int* __restrict__ counts)
{
  int t    = (blockIdx.x * blockDim.x + threadIdx.x) >> 5;
  int lane = threadIdx.x & 31;
  if (t >= T_TOK) return;
  float acc[NE];
  #pragma unroll
  for (int e = 0; e < NE; ++e) acc[e] = 0.f;
  const float* xr = X + (size_t)t * HID;
  for (int h = lane; h < HID; h += 32) {
    float xv = xr[h];
    const float* wr = Wg + (size_t)h * NE;
    #pragma unroll
    for (int e = 0; e < NE; ++e) acc[e] += xv * wr[e];
  }
  #pragma unroll
  for (int off = 16; off > 0; off >>= 1) {
    #pragma unroll
    for (int e = 0; e < NE; ++e) acc[e] += __shfl_xor(acc[e], off, 32);
  }
  // top-2 (first-index wins on ties, matching jax top_k)
  int i0 = 0; float l0 = acc[0];
  #pragma unroll
  for (int e = 1; e < NE; ++e) if (acc[e] > l0) { l0 = acc[e]; i0 = e; }
  int i1 = (i0 == 0) ? 1 : 0; float l1 = acc[i1];
  #pragma unroll
  for (int e = 0; e < NE; ++e) if (e != i0 && acc[e] > l1) { l1 = acc[e]; i1 = e; }
  // softmax-all + top2 + renorm == softmax over the top-2 logits
  float w0 = 1.f / (1.f + __expf(l1 - l0));
  if (lane == 0) {
    tok_e[2*t]   = i0; tok_w[2*t]   = w0;
    tok_e[2*t+1] = i1; tok_w[2*t+1] = 1.f - w0;
    atomicAdd(&counts[i0], 1);
    atomicAdd(&counts[i1], 1);
  }
}

__global__ void scan_kernel(const int* __restrict__ counts, int* __restrict__ offs) {
  if (threadIdx.x == 0) {
    int s = 0;
    for (int e = 0; e < NE; ++e) { offs[e] = s; s += counts[e]; }
    offs[NE] = s;
  }
}

__global__ __launch_bounds__(256) void assign_kernel(
    const int* __restrict__ tok_e, int* __restrict__ cursors,
    const int* __restrict__ offs, int* __restrict__ slot_token,
    int* __restrict__ tok_slot)
{
  int t = blockIdx.x * blockDim.x + threadIdx.x;
  if (t >= T_TOK) return;
  #pragma unroll
  for (int k = 0; k < TOPK; ++k) {
    int e    = tok_e[2*t+k];
    int pos  = atomicAdd(&cursors[e], 1);
    int slot = offs[e] + pos;
    slot_token[slot] = t;
    tok_slot[2*t+k]  = slot;
  }
}

// ------------------------------------------------- gate_up GEMM + SiLU*mul
// Block: act[:, n0:n0+128] = silu(X@Wgate[:,n0:+128]) * (X@Wup[:,n0:+128])
// 8 waves: wave (wm, wn) owns rows wm*16..+16, cols wn*64..+64 (4 subtiles),
// i.e. 8 WMMAs (4 gate + 4 up) per A-fragment load.
template <bool GATHER>
__global__ __launch_bounds__(256) void gateup_silu_kernel(
    const float* __restrict__ X, const float* __restrict__ WguBase,
    _Float16* __restrict__ Act,
    const int* __restrict__ gather, const int* __restrict__ offs,
    int Mfull, int64_t strideB)
{
  int e = blockIdx.z;
  int r0 = 0, cnt = Mfull;
  if (offs) { r0 = offs[e]; cnt = offs[e+1] - r0; }
  int m0 = blockIdx.y * BM;
  if (m0 >= cnt) return;
  int n0 = blockIdx.x * BN;
  const float* B = WguBase + (int64_t)e * strideB;

  __shared__ __align__(16) _Float16 As[BM*LDA];
  __shared__ __align__(16) _Float16 Bg[BN*LDB];
  __shared__ __align__(16) _Float16 Bu[BN*LDB];

  int tid  = threadIdx.x;
  int lane = tid & 31;
  int wid  = tid >> 5;
  int wm   = wid & 3;
  int wn   = wid >> 2;
  int kh   = lane >> 4;   // K-half held by this lane
  int rl   = lane & 15;   // row (A) / col (B,C) within 16x16 tile

  v8f accg[4] = {};
  v8f accu[4] = {};

  for (int k0 = 0; k0 < HID; k0 += BK) {
    __syncthreads();
    // A tile: 64x32 f32 -> f16 LDS. Branchless: clamp row (epilogue guards M).
    #pragma unroll
    for (int ii = 0; ii < (BM*(BK/4))/256; ++ii) {      // 2 iters
      int i = tid + ii*256;
      int r = i >> 3, c4 = i & 7;
      int gr = m0 + r; if (gr > cnt - 1) gr = cnt - 1;
      int row = GATHER ? gather[r0 + gr] : (r0 + gr);
      float4 f = *(const float4*)(X + (size_t)row * HID + k0 + c4*4);
      h4 hv; hv.x=(_Float16)f.x; hv.y=(_Float16)f.y; hv.z=(_Float16)f.z; hv.w=(_Float16)f.w;
      *(h4*)(As + r*LDA + c4*4) = hv;
    }
    // B tiles (gate + up): two adjacent k rows per thread, packed half2 stores
    #pragma unroll
    for (int ii = 0; ii < ((BK/2)*(BN/4))/256; ++ii) {  // 2 iters
      int i  = tid + ii*256;
      int k2 = (i >> 5) * 2;          // even k
      int n  = (i & 31) * 4;
      const float* bp0 = B + (int64_t)(k0 + k2) * (2*IEXP) + n0 + n;
      const float* bp1 = bp0 + (2*IEXP);
      float4 g0 = *(const float4*)(bp0);
      float4 g1 = *(const float4*)(bp1);
      float4 u0 = *(const float4*)(bp0 + IEXP);
      float4 u1 = *(const float4*)(bp1 + IEXP);
      h2 p;
      p.x=(_Float16)g0.x; p.y=(_Float16)g1.x; *(h2*)(Bg+(n+0)*LDB+k2)=p;
      p.x=(_Float16)g0.y; p.y=(_Float16)g1.y; *(h2*)(Bg+(n+1)*LDB+k2)=p;
      p.x=(_Float16)g0.z; p.y=(_Float16)g1.z; *(h2*)(Bg+(n+2)*LDB+k2)=p;
      p.x=(_Float16)g0.w; p.y=(_Float16)g1.w; *(h2*)(Bg+(n+3)*LDB+k2)=p;
      p.x=(_Float16)u0.x; p.y=(_Float16)u1.x; *(h2*)(Bu+(n+0)*LDB+k2)=p;
      p.x=(_Float16)u0.y; p.y=(_Float16)u1.y; *(h2*)(Bu+(n+1)*LDB+k2)=p;
      p.x=(_Float16)u0.z; p.y=(_Float16)u1.z; *(h2*)(Bu+(n+2)*LDB+k2)=p;
      p.x=(_Float16)u0.w; p.y=(_Float16)u1.w; *(h2*)(Bu+(n+3)*LDB+k2)=p;
    }
    __syncthreads();

    // A frag: lane<16 -> K {0..15}; lane>=16 -> K {16..31}
    FragH a;
    const uint4* pa = (const uint4*)(As + (wm*16 + rl)*LDA + kh*16);
    a.q[0] = pa[0]; a.q[1] = pa[1];
    #pragma unroll
    for (int tI = 0; tI < 4; ++tI) {
      int col = wn*64 + tI*16 + rl;
      FragH bg, bu;
      const uint4* pg = (const uint4*)(Bg + col*LDB + kh*16);
      bg.q[0] = pg[0]; bg.q[1] = pg[1];
      const uint4* pu = (const uint4*)(Bu + col*LDB + kh*16);
      bu.q[0] = pu[0]; bu.q[1] = pu[1];
      accg[tI] = __builtin_amdgcn_wmma_f32_16x16x32_f16(false, a.v, false, bg.v,
                                                        (short)0, accg[tI], false, false);
      accu[tI] = __builtin_amdgcn_wmma_f32_16x16x32_f16(false, a.v, false, bu.v,
                                                        (short)0, accu[tI], false, false);
    }
  }

  // Epilogue: guard once per output row (r depends on v only, not tI)
  #pragma unroll
  for (int v = 0; v < 8; ++v) {
    int r = m0 + wm*16 + kh*8 + v;   // C layout: lanes>=16 -> rows 8..15
    if (r < cnt) {
      _Float16* arow = Act + (size_t)(r0 + r) * IEXP + n0 + wn*64 + rl;
      #pragma unroll
      for (int tI = 0; tI < 4; ++tI) {
        float g = accg[tI][v], u = accu[tI][v];
        float s = g / (1.f + __expf(-g));
        arow[tI*16] = (_Float16)(s * u);
      }
    }
  }
}

// ------------------------------------------------------------ down GEMM
// Block: C[:, n0:n0+128] = Act @ Wd; wave = 16x64 (4 WMMAs per A fragment)
__global__ __launch_bounds__(256) void down_kernel(
    const _Float16* __restrict__ ActIn, const float* __restrict__ WdBase,
    float* __restrict__ Cout,
    const int* __restrict__ offs, int Mfull, int64_t strideB)
{
  int e = blockIdx.z;
  int r0 = 0, cnt = Mfull;
  if (offs) { r0 = offs[e]; cnt = offs[e+1] - r0; }
  int m0 = blockIdx.y * BM;
  if (m0 >= cnt) return;
  int n0 = blockIdx.x * BN;
  const float* B = WdBase + (int64_t)e * strideB;

  __shared__ __align__(16) _Float16 As[BM*LDA];
  __shared__ __align__(16) _Float16 Bs[BN*LDB];

  int tid  = threadIdx.x;
  int lane = tid & 31;
  int wid  = tid >> 5;
  int wm   = wid & 3;
  int wn   = wid >> 2;
  int kh   = lane >> 4;
  int rl   = lane & 15;

  v8f acc[4] = {};

  for (int k0 = 0; k0 < IEXP; k0 += BK) {
    __syncthreads();
    {   // A tile: f16 already; 64 rows x 32 halves = 256 b128 ops, clamped rows
      int r = tid >> 2, c8 = tid & 3;
      int gr = m0 + r; if (gr > cnt - 1) gr = cnt - 1;
      uint4 q = *(const uint4*)(ActIn + (size_t)(r0 + gr) * IEXP + k0 + c8*8);
      *(uint4*)(As + r*LDA + c8*8) = q;
    }
    #pragma unroll
    for (int ii = 0; ii < ((BK/2)*(BN/4))/256; ++ii) {  // 2 iters
      int i  = tid + ii*256;
      int k2 = (i >> 5) * 2;
      int n  = (i & 31) * 4;
      const float* bp0 = B + (int64_t)(k0 + k2) * HID + n0 + n;
      float4 f0 = *(const float4*)(bp0);
      float4 f1 = *(const float4*)(bp0 + HID);
      h2 p;
      p.x=(_Float16)f0.x; p.y=(_Float16)f1.x; *(h2*)(Bs+(n+0)*LDB+k2)=p;
      p.x=(_Float16)f0.y; p.y=(_Float16)f1.y; *(h2*)(Bs+(n+1)*LDB+k2)=p;
      p.x=(_Float16)f0.z; p.y=(_Float16)f1.z; *(h2*)(Bs+(n+2)*LDB+k2)=p;
      p.x=(_Float16)f0.w; p.y=(_Float16)f1.w; *(h2*)(Bs+(n+3)*LDB+k2)=p;
    }
    __syncthreads();

    FragH a;
    const uint4* pa = (const uint4*)(As + (wm*16 + rl)*LDA + kh*16);
    a.q[0] = pa[0]; a.q[1] = pa[1];
    #pragma unroll
    for (int tI = 0; tI < 4; ++tI) {
      int col = wn*64 + tI*16 + rl;
      FragH b;
      const uint4* pb = (const uint4*)(Bs + col*LDB + kh*16);
      b.q[0] = pb[0]; b.q[1] = pb[1];
      acc[tI] = __builtin_amdgcn_wmma_f32_16x16x32_f16(false, a.v, false, b.v,
                                                       (short)0, acc[tI], false, false);
    }
  }

  #pragma unroll
  for (int v = 0; v < 8; ++v) {
    int r = m0 + wm*16 + kh*8 + v;
    if (r < cnt) {
      float* crow = Cout + (size_t)(r0 + r) * HID + n0 + wn*64 + rl;
      #pragma unroll
      for (int tI = 0; tI < 4; ++tI) crow[tI*16] = acc[tI][v];
    }
  }
}

// ------------------------------------------------------------- combine
__global__ __launch_bounds__(256) void combine_kernel(
    float* __restrict__ out, const float* __restrict__ slot_out,
    const int* __restrict__ tok_slot, const float* __restrict__ tok_w)
{
  int idx = blockIdx.x * blockDim.x + threadIdx.x;   // T*H/4 threads
  int t  = idx >> 8;          // H/4 = 256 float4 per token
  int hc = idx & 255;
  float4 o = ((const float4*)out)[idx];              // shared-expert result
  #pragma unroll
  for (int k = 0; k < TOPK; ++k) {
    int   s = tok_slot[2*t + k];
    float w = tok_w[2*t + k];
    float4 v = ((const float4*)(slot_out + (size_t)s * HID))[hc];
    o.x += w * v.x; o.y += w * v.y; o.z += w * v.z; o.w += w * v.w;
  }
  ((float4*)out)[idx] = o;
}

// ------------------------------------------------------------- launcher
extern "C" void kernel_launch(void* const* d_in, const int* in_sizes, int n_in,
                              void* d_out, int out_size, void* d_ws, size_t ws_size,
                              hipStream_t stream) {
  (void)in_sizes; (void)n_in; (void)out_size; (void)ws_size;
  const float* X    = (const float*)d_in[0];
  const float* Wg   = (const float*)d_in[1];
  const float* Wegu = (const float*)d_in[2];
  const float* Wed  = (const float*)d_in[3];
  const float* Wsgu = (const float*)d_in[4];
  const float* Wsd  = (const float*)d_in[5];
  float* out = (float*)d_out;

  char* ws = (char*)d_ws;
  int* counts  = (int*)ws;        // [8]
  int* cursors = counts + 8;      // [8]
  int* offs    = counts + 16;     // [9]
  char* p = ws + 256;
  int*      tok_e    = (int*)p;      p += (size_t)NSLOT * 4;
  float*    tok_w    = (float*)p;    p += (size_t)NSLOT * 4;
  int*      tok_slot = (int*)p;      p += (size_t)NSLOT * 4;
  int*      slot_tok = (int*)p;      p += (size_t)NSLOT * 4;
  _Float16* shact    = (_Float16*)p; p += (size_t)T_TOK * IEXP * 2;   // 8 MB
  _Float16* act      = (_Float16*)p; p += (size_t)NSLOT * IEXP * 2;   // 16 MB
  float*    slot_out = (float*)p;                                     // 64 MB

  hipMemsetAsync(counts, 0, 64, stream);   // counts + cursors

  router_kernel<<<T_TOK/8, 256, 0, stream>>>(X, Wg, tok_e, tok_w, counts);
  scan_kernel<<<1, 32, 0, stream>>>(counts, offs);
  assign_kernel<<<T_TOK/256, 256, 0, stream>>>(tok_e, cursors, offs, slot_tok, tok_slot);

  // shared expert: X -> shact -> out
  gateup_silu_kernel<false><<<dim3(IEXP/BN, T_TOK/BM, 1), 256, 0, stream>>>(
      X, Wsgu, shact, nullptr, nullptr, T_TOK, 0);
  down_kernel<<<dim3(HID/BN, T_TOK/BM, 1), 256, 0, stream>>>(
      shact, Wsd, out, nullptr, T_TOK, 0);

  // routed experts: gathered grouped GEMMs over compacted slot lists
  gateup_silu_kernel<true><<<dim3(IEXP/BN, T_TOK/BM, NE), 256, 0, stream>>>(
      X, Wegu, act, slot_tok, offs, 0, (int64_t)HID * (2*IEXP));
  down_kernel<<<dim3(HID/BN, T_TOK/BM, NE), 256, 0, stream>>>(
      act, Wed, slot_out, offs, 0, (int64_t)IEXP * HID);

  combine_kernel<<<(T_TOK*(HID/4))/256, 256, 0, stream>>>(out, slot_out, tok_slot, tok_w);
}
